// CausalSelfAttention_82918638616940
// MI455X (gfx1250) — compile-verified, matrix operand order checked by reading, not run on hardware
//
#include <hip/hip_runtime.h>

typedef unsigned short bf16u;
typedef __attribute__((ext_vector_type(16))) __bf16 v16bf;
typedef __attribute__((ext_vector_type(8)))  float  v8f;

#define TSEQ   4096
#define NBATCH 2
#define NHEAD  12
#define DHEAD  64
#define CMODEL 768

__device__ __forceinline__ bf16u f32_to_bf16(float f) {
  union { float f; unsigned int u; } v; v.f = f;
  unsigned int r = 0x7FFFu + ((v.u >> 16) & 1u);
  return (bf16u)((v.u + r) >> 16);
}

// Load a 16-element bf16 fragment (A/B layout for v_wmma_*_16x16x32):
// elements h=0..7 at base[koff + h], h=8..15 at base[koff + 16 + h-8].
__device__ __forceinline__ v16bf frag_ld(const bf16u* base, int koff) {
  union { uint4 q[2]; v16bf v; } u;
  u.q[0] = *(const uint4*)(base + koff);
  u.q[1] = *(const uint4*)(base + koff + 16);
  return u.v;
}

__device__ __forceinline__ v8f wmma_bf16(v16bf a, v16bf b, v8f c) {
  return __builtin_amdgcn_wmma_f32_16x16x32_bf16(
      false, a, false, b, (short)0, c, false, false);
}

__device__ __forceinline__ v8f v8f_zero() {
  v8f v;
#pragma unroll
  for (int i = 0; i < 8; ++i) v[i] = 0.0f;
  return v;
}

// Async global->LDS 16-byte copy (CDNA5 GLOBAL_LOAD_ASYNC_TO_LDS, GVS mode:
// mem = SADDR64 + VADDR32 + 0; LDS dst = low 32 bits of generic shared ptr).
// Tracked on ASYNCcnt.
__device__ __forceinline__ void async_ld_b128(void* lds_ptr, const void* gbase,
                                              unsigned byte_off) {
  unsigned l = (unsigned)(size_t)lds_ptr;
  asm volatile("global_load_async_to_lds_b128 %0, %1, %2"
               :: "v"(l), "v"(byte_off), "s"(gbase)
               : "memory");
}
__device__ __forceinline__ void wait_async0() {
  asm volatile("s_wait_asynccnt 0x0" ::: "memory");
}

// ---------------------------------------------------------------------------
// Kernel 1: qkv = x @ W_qkv + b_qkv, scattered to head-major bf16 Q/K/V.
// Block tile 128x128, 8 waves, K-step 32. (f32 inputs need conversion, so
// staging stays through VGPRs here.)
// ---------------------------------------------------------------------------
__global__ __launch_bounds__(256) void qkv_gemm_kernel(
    const float* __restrict__ x, const float* __restrict__ Wqkv,
    const float* __restrict__ bqkv, bf16u* __restrict__ qkvws) {
  __shared__ bf16u As[128 * 32];   // [m][k] row-major
  __shared__ bf16u Bs[128 * 32];   // [n][k] (transposed) row-major
  const int t    = threadIdx.x;
  const int wid  = t >> 5;
  const int lane = t & 31;
  const int half = lane >> 4;
  const int l16  = lane & 15;
  const int koff = half * 8;
  const int mbase = blockIdx.x * 128;
  const int nbase = blockIdx.y * 128;

  v8f acc[8];
#pragma unroll
  for (int nb = 0; nb < 8; ++nb) acc[nb] = v8f_zero();

  for (int k0 = 0; k0 < CMODEL; k0 += 32) {
#pragma unroll
    for (int i = 0; i < 4; ++i) {                // A: 128x32 f32 -> bf16
      int j = t + i * 256;                       // float4 id 0..1023
      int row = j >> 3;
      int kq  = (j & 7) * 4;
      float4 f = *(const float4*)(x + (size_t)(mbase + row) * CMODEL + k0 + kq);
      uint2 p;
      p.x = (unsigned)f32_to_bf16(f.x) | ((unsigned)f32_to_bf16(f.y) << 16);
      p.y = (unsigned)f32_to_bf16(f.z) | ((unsigned)f32_to_bf16(f.w) << 16);
      *(uint2*)(As + row * 32 + kq) = p;
    }
#pragma unroll
    for (int i = 0; i < 4; ++i) {                // B: 32x128 -> Bs[n][k]
      int j = t + i * 256;
      int kr = j >> 5;
      int nq = (j & 31) * 4;
      float4 f = *(const float4*)(Wqkv + (size_t)(k0 + kr) * (3 * CMODEL) + nbase + nq);
      Bs[(nq + 0) * 32 + kr] = f32_to_bf16(f.x);
      Bs[(nq + 1) * 32 + kr] = f32_to_bf16(f.y);
      Bs[(nq + 2) * 32 + kr] = f32_to_bf16(f.z);
      Bs[(nq + 3) * 32 + kr] = f32_to_bf16(f.w);
    }
    __syncthreads();
    v16bf a = frag_ld(As + (wid * 16 + l16) * 32, koff);
#pragma unroll
    for (int nb = 0; nb < 8; ++nb) {
      v16bf bfr = frag_ld(Bs + (nb * 16 + l16) * 32, koff);
      acc[nb] = wmma_bf16(a, bfr, acc[nb]);
    }
    __syncthreads();
  }

  const size_t QSZ = (size_t)NBATCH * NHEAD * TSEQ * DHEAD;
#pragma unroll
  for (int nb = 0; nb < 8; ++nb) {
    int gc = nbase + nb * 16 + l16;
    float bias = bqkv[gc];
    int which = gc / CMODEL;          // 0=q, 1=k, 2=v
    int cc = gc - which * CMODEL;
    int hh = cc >> 6;
    int dd = cc & 63;
    bf16u* dst = qkvws + (size_t)which * QSZ;
#pragma unroll
    for (int r = 0; r < 8; ++r) {
      int gm = mbase + wid * 16 + r + 8 * half;
      int bb = gm >> 12;              // / 4096
      int trow = gm & 4095;
      dst[(((size_t)bb * NHEAD + hh) * TSEQ + trow) * DHEAD + dd] =
          f32_to_bf16(acc[nb][r] + bias);
    }
  }
}

// ---------------------------------------------------------------------------
// Kernel 2: causal flash attention. Grid (B*H, T/64), 4 waves; each wave
// owns a 16x64 Q tile, streams K/V in 32-key tiles through LDS.
// K tile staged with async global->LDS DMA; V transposed through registers.
// ---------------------------------------------------------------------------
__global__ __launch_bounds__(128) void flash_attn_kernel(
    const bf16u* __restrict__ Qw, const bf16u* __restrict__ Kw,
    const bf16u* __restrict__ Vw, bf16u* __restrict__ attn) {
  __shared__ bf16u Klds[32 * 64];     // [key][d]
  __shared__ bf16u Vtlds[64 * 32];    // [d][key]  (transposed)
  __shared__ bf16u Plds[4 * 16 * 32]; // per-wave [m][key]
  const int t    = threadIdx.x;
  const int wid  = t >> 5;
  const int lane = t & 31;
  const int half = lane >> 4;
  const int l16  = lane & 15;
  const int koff = half * 8;
  const int bh   = blockIdx.x;
  const int b    = bh / NHEAD;
  const int h    = bh % NHEAD;
  const int qblk = blockIdx.y;
  const int qr0  = qblk * 64 + wid * 16;
  const bf16u* Qbh = Qw + (size_t)bh * TSEQ * DHEAD;
  const bf16u* Kbh = Kw + (size_t)bh * TSEQ * DHEAD;
  const bf16u* Vbh = Vw + (size_t)bh * TSEQ * DHEAD;

  // Q A-fragments (d-chunks 0..31 and 32..63), loaded once from global.
  v16bf qa0 = frag_ld(Qbh + (size_t)(qr0 + l16) * DHEAD + 0, koff);
  v16bf qa1 = frag_ld(Qbh + (size_t)(qr0 + l16) * DHEAD + 32, koff);

  v8f o0 = v8f_zero(), o1 = v8f_zero(), o2 = v8f_zero(), o3 = v8f_zero();
  float rowmax[8], rowsum[8];
#pragma unroll
  for (int r = 0; r < 8; ++r) { rowmax[r] = -3.0e38f; rowsum[r] = 0.0f; }

  const float cs = 0.125f * 1.44269504f;   // 1/sqrt(64) * log2(e)
  const int ntiles = qblk * 2 + 2;
  bf16u* Pw = Plds + wid * (16 * 32);

  for (int kb = 0; kb < ntiles; ++kb) {
    const int kb0 = kb * 32;
    {  // K tile: async DMA global->LDS (4KB, straight copy)
      unsigned gbyte = (unsigned)((size_t)kb0 * DHEAD * sizeof(bf16u));
      async_ld_b128((char*)Klds + t * 16,        Kbh, gbyte + t * 16);
      async_ld_b128((char*)Klds + 2048 + t * 16, Kbh, gbyte + 2048 + t * 16);
      // V tile: load + transpose into Vtlds[d][key]
      const uint4* vs = (const uint4*)(Vbh + (size_t)kb0 * DHEAD);
#pragma unroll
      for (int i = 0; i < 2; ++i) {
        int j = t + i * 128;
        int key = j >> 3;
        int dq  = (j & 7) * 8;
        union { uint4 q; bf16u s[8]; } u;
        u.q = vs[j];
#pragma unroll
        for (int c = 0; c < 8; ++c) Vtlds[(dq + c) * 32 + key] = u.s[c];
      }
      // prefetch next K/V tiles into cache while this one is consumed
      if (kb + 1 < ntiles) {
        const bf16u* nk = Kbh + (size_t)(kb0 + 32) * DHEAD;
        const bf16u* nv = Vbh + (size_t)(kb0 + 32) * DHEAD;
        __builtin_prefetch(nk + t * 16, 0, 0);
        __builtin_prefetch(nv + t * 16, 0, 0);
      }
      wait_async0();
    }
    __syncthreads();
    if (kb0 <= qr0 + 15) {   // tile not fully masked for this wave
      v8f Sa = v8f_zero(), Sb = v8f_zero();
      {
        v16bf bk;
        bk = frag_ld(Klds + l16 * 64 + 0, koff);         Sa = wmma_bf16(qa0, bk, Sa);
        bk = frag_ld(Klds + l16 * 64 + 32, koff);        Sa = wmma_bf16(qa1, bk, Sa);
        bk = frag_ld(Klds + (l16 + 16) * 64 + 0, koff);  Sb = wmma_bf16(qa0, bk, Sb);
        bk = frag_ld(Klds + (l16 + 16) * 64 + 32, koff); Sb = wmma_bf16(qa1, bk, Sb);
      }
      float sa[8], sb[8];
#pragma unroll
      for (int r = 0; r < 8; ++r) { sa[r] = Sa[r] * cs; sb[r] = Sb[r] * cs; }
      if (kb0 + 31 > qr0) {  // causal mask needed
#pragma unroll
        for (int r = 0; r < 8; ++r) {
          int q = qr0 + r + 8 * half;
          if (kb0 + l16 > q)      sa[r] = -3.0e38f;
          if (kb0 + 16 + l16 > q) sb[r] = -3.0e38f;
        }
      }
      float tmax[8];
#pragma unroll
      for (int r = 0; r < 8; ++r) tmax[r] = fmaxf(sa[r], sb[r]);
#pragma unroll
      for (int ofs = 1; ofs < 16; ofs <<= 1) {
#pragma unroll
        for (int r = 0; r < 8; ++r)
          tmax[r] = fmaxf(tmax[r], __shfl_xor(tmax[r], ofs, 16));
      }
      float sc8[8];
#pragma unroll
      for (int r = 0; r < 8; ++r) {
        float nm = fmaxf(rowmax[r], tmax[r]);
        sc8[r] = exp2f(rowmax[r] - nm);
        rowmax[r] = nm;
      }
#pragma unroll
      for (int r = 0; r < 8; ++r) {
        o0[r] *= sc8[r]; o1[r] *= sc8[r]; o2[r] *= sc8[r]; o3[r] *= sc8[r];
      }
      float part[8];
#pragma unroll
      for (int r = 0; r < 8; ++r) {
        float pa = exp2f(sa[r] - rowmax[r]);
        float pb = exp2f(sb[r] - rowmax[r]);
        part[r] = pa + pb;
        Pw[(r + 8 * half) * 32 + l16]      = f32_to_bf16(pa);
        Pw[(r + 8 * half) * 32 + l16 + 16] = f32_to_bf16(pb);
      }
#pragma unroll
      for (int ofs = 1; ofs < 16; ofs <<= 1) {
#pragma unroll
        for (int r = 0; r < 8; ++r) part[r] += __shfl_xor(part[r], ofs, 16);
      }
#pragma unroll
      for (int r = 0; r < 8; ++r) rowsum[r] = rowsum[r] * sc8[r] + part[r];
      // P written by this wave, read by this wave: LDS is in-order per wave,
      // but force the compiler not to reorder across the store->load.
      asm volatile("s_wait_dscnt 0x0" ::: "memory");
      v16bf pf = frag_ld(Pw + l16 * 32, koff);
      o0 = wmma_bf16(pf, frag_ld(Vtlds + (0 * 16 + l16) * 32, koff), o0);
      o1 = wmma_bf16(pf, frag_ld(Vtlds + (1 * 16 + l16) * 32, koff), o1);
      o2 = wmma_bf16(pf, frag_ld(Vtlds + (2 * 16 + l16) * 32, koff), o2);
      o3 = wmma_bf16(pf, frag_ld(Vtlds + (3 * 16 + l16) * 32, koff), o3);
    }
    __syncthreads();
  }

  float inv[8];
#pragma unroll
  for (int r = 0; r < 8; ++r) inv[r] = 1.0f / rowsum[r];
#pragma unroll
  for (int r = 0; r < 8; ++r) {
    int q = qr0 + r + 8 * half;
    bf16u* arow = attn + ((size_t)b * TSEQ + q) * CMODEL + h * DHEAD;
    arow[0 * 16 + l16] = f32_to_bf16(o0[r] * inv[r]);
    arow[1 * 16 + l16] = f32_to_bf16(o1[r] * inv[r]);
    arow[2 * 16 + l16] = f32_to_bf16(o2[r] * inv[r]);
    arow[3 * 16 + l16] = f32_to_bf16(o3[r] * inv[r]);
  }
}

// ---------------------------------------------------------------------------
// Kernel 3: out = attn @ W_out + b_out (f32 output).
// bf16 A tile staged via async global->LDS DMA.
// ---------------------------------------------------------------------------
__global__ __launch_bounds__(256) void out_gemm_kernel(
    const bf16u* __restrict__ attn, const float* __restrict__ Wout,
    const float* __restrict__ bout, float* __restrict__ out) {
  __shared__ bf16u As[128 * 32];
  __shared__ bf16u Bs[128 * 32];
  const int t    = threadIdx.x;
  const int wid  = t >> 5;
  const int lane = t & 31;
  const int half = lane >> 4;
  const int l16  = lane & 15;
  const int koff = half * 8;
  const int mbase = blockIdx.x * 128;
  const int nbase = blockIdx.y * 128;

  v8f acc[8];
#pragma unroll
  for (int nb = 0; nb < 8; ++nb) acc[nb] = v8f_zero();

  for (int k0 = 0; k0 < CMODEL; k0 += 32) {
#pragma unroll
    for (int i = 0; i < 2; ++i) {      // A: bf16 async copy, 512 x b128
      int j = t + i * 256;
      int row = j >> 2;
      int kq  = (j & 3) * 8;
      async_ld_b128(As + row * 32 + kq, attn,
                    (unsigned)(((size_t)(mbase + row) * CMODEL + k0 + kq) *
                               sizeof(bf16u)));
    }
#pragma unroll
    for (int i = 0; i < 4; ++i) {      // B -> Bs[n][k]
      int j = t + i * 256;
      int kr = j >> 5;
      int nq = (j & 31) * 4;
      float4 f = *(const float4*)(Wout + (size_t)(k0 + kr) * CMODEL + nbase + nq);
      Bs[(nq + 0) * 32 + kr] = f32_to_bf16(f.x);
      Bs[(nq + 1) * 32 + kr] = f32_to_bf16(f.y);
      Bs[(nq + 2) * 32 + kr] = f32_to_bf16(f.z);
      Bs[(nq + 3) * 32 + kr] = f32_to_bf16(f.w);
    }
    wait_async0();
    __syncthreads();
    v16bf a = frag_ld(As + (wid * 16 + l16) * 32, koff);
#pragma unroll
    for (int nb = 0; nb < 8; ++nb) {
      v16bf bfr = frag_ld(Bs + (nb * 16 + l16) * 32, koff);
      acc[nb] = wmma_bf16(a, bfr, acc[nb]);
    }
    __syncthreads();
  }
#pragma unroll
  for (int nb = 0; nb < 8; ++nb) {
    int gc = nbase + nb * 16 + l16;
    float bias = bout[gc];
#pragma unroll
    for (int r = 0; r < 8; ++r) {
      int gm = mbase + wid * 16 + r + 8 * half;
      out[(size_t)gm * CMODEL + gc] = acc[nb][r] + bias;
    }
  }
}

// ---------------------------------------------------------------------------
extern "C" void kernel_launch(void* const* d_in, const int* in_sizes, int n_in,
                              void* d_out, int out_size, void* d_ws, size_t ws_size,
                              hipStream_t stream) {
  (void)in_sizes; (void)n_in; (void)out_size; (void)ws_size;
  const float* x    = (const float*)d_in[0];
  const float* Wqkv = (const float*)d_in[1];
  const float* bqkv = (const float*)d_in[2];
  const float* Wout = (const float*)d_in[3];
  const float* bout = (const float*)d_in[4];

  bf16u* Qw = (bf16u*)d_ws;
  const size_t QSZ = (size_t)NBATCH * NHEAD * TSEQ * DHEAD;  // 6.29M elems
  bf16u* Kw   = Qw + QSZ;
  bf16u* Vw   = Kw + QSZ;
  bf16u* attn = Vw + QSZ;        // [B*T, 768] bf16
  float* out  = (float*)d_out;

  qkv_gemm_kernel<<<dim3((NBATCH * TSEQ) / 128, (3 * CMODEL) / 128), 256, 0, stream>>>(
      x, Wqkv, bqkv, Qw);
  flash_attn_kernel<<<dim3(NBATCH * NHEAD, TSEQ / 64), 128, 0, stream>>>(
      Qw, Kw, Vw, attn);
  out_gemm_kernel<<<dim3((NBATCH * TSEQ) / 128, CMODEL / 128), 256, 0, stream>>>(
      attn, Wout, bout, out);
}
